// ConvMambaBlock_57415122813610
// MI455X (gfx1250) — matrix operand order repaired and checked by reading, not compiled, not run on hardware
//
#include <hip/hip_runtime.h>
#include <math.h>

typedef __attribute__((ext_vector_type(16))) _Float16 v16h;
typedef __attribute__((ext_vector_type(8)))  float    v8f;
typedef __attribute__((ext_vector_type(4)))  unsigned v4u;

#define BATCH 4
#define SEQL  1024
#define DIMC  256
#define DINNER 512
#define DSTATE 32
#define DTRANK 16
#define MROWS (BATCH * SEQL)   // 4096
#define XDBLS 96               // x_dbl row stride (80 -> 96)
#define WXPROWS 128            // x_proj weight padded rows (80 -> 128)

// ---------------------------------------------------------------------------
// f32 -> f16 weight conversion (run once per launch; weights live in L2)
// ---------------------------------------------------------------------------
__global__ void cvt_f32_to_f16(const float* __restrict__ src, _Float16* __restrict__ dst, int n) {
    int i = blockIdx.x * blockDim.x + threadIdx.x;
    if (i < n) dst[i] = (_Float16)src[i];
}

// x_proj (80,512) -> zero-padded (128,512) f16 so GEMM loads need no clamping
__global__ void cvt_pad_xproj(const float* __restrict__ src, _Float16* __restrict__ dst) {
    int i = blockIdx.x * blockDim.x + threadIdx.x;   // over 128*512
    int nrow = i / DINNER;
    int k    = i - nrow * DINNER;
    dst[i] = (nrow < 80) ? (_Float16)src[nrow * DINNER + k] : (_Float16)0.0f;
}

// ---------------------------------------------------------------------------
// LayerNorm over last dim (256). One block per row, tree reduction in LDS.
// ---------------------------------------------------------------------------
__global__ void ln_kernel(const float* __restrict__ x, const float* __restrict__ g,
                          const float* __restrict__ b, float* __restrict__ outF,
                          _Float16* __restrict__ outH) {
    __shared__ float red[DIMC];
    const int m = blockIdx.x, c = threadIdx.x;
    const size_t idx = (size_t)m * DIMC + c;
    float v = x[idx];
    red[c] = v; __syncthreads();
    for (int s = DIMC / 2; s > 0; s >>= 1) { if (c < s) red[c] += red[c + s]; __syncthreads(); }
    float mu = red[0] * (1.0f / DIMC); __syncthreads();
    float d = v - mu;
    red[c] = d * d; __syncthreads();
    for (int s = DIMC / 2; s > 0; s >>= 1) { if (c < s) red[c] += red[c + s]; __syncthreads(); }
    float var = red[0] * (1.0f / DIMC);
    float y = d * rsqrtf(var + 1e-5f) * g[c] + b[c];
    if (outF) outF[idx] = y;
    if (outH) outH[idx] = (_Float16)y;
}

// ---------------------------------------------------------------------------
// Local depthwise conv (K=3, pad 1,1) over L, plus residual xn: u0 = conv(xn)+xn
// ---------------------------------------------------------------------------
__global__ void conv_local_add(const float* __restrict__ xn, const float* __restrict__ w,
                               const float* __restrict__ bc, _Float16* __restrict__ u0) {
    const int m = blockIdx.x, c = threadIdx.x;
    const int bidx = m >> 10, l = m & (SEQL - 1);
    float acc = xn[(size_t)m * DIMC + c] + bc[c];
    #pragma unroll
    for (int k = 0; k < 3; ++k) {
        int ll = l - 1 + k;
        if (ll >= 0 && ll < SEQL)
            acc += xn[((size_t)(bidx << 10) + ll) * DIMC + c] * w[c * 3 + k];
    }
    u0[(size_t)m * DIMC + c] = (_Float16)acc;
}

// ---------------------------------------------------------------------------
// Mamba causal depthwise conv (K=4, pad_left=3) + bias + SiLU on xin = xz[:, :512]
// ---------------------------------------------------------------------------
__global__ void conv_mamba_silu(const float* __restrict__ xz, const float* __restrict__ w,
                                const float* __restrict__ bc, _Float16* __restrict__ o16,
                                float* __restrict__ o32) {
    const int m = blockIdx.x, d = threadIdx.x;
    const int bidx = m >> 10, l = m & (SEQL - 1);
    float acc = bc[d];
    #pragma unroll
    for (int k = 0; k < 4; ++k) {
        int ll = l - 3 + k;
        if (ll >= 0)
            acc += xz[((size_t)(bidx << 10) + ll) * (2 * DINNER) + d] * w[d * 4 + k];
    }
    float s = acc / (1.0f + __expf(-acc));
    o16[(size_t)m * DINNER + d] = (_Float16)s;
    o32[(size_t)m * DINNER + d] = s;
}

// ---------------------------------------------------------------------------
// delta = softplus(dt @ dt_w.T + dt_b), dt = x_dbl[:, :16] (stride XDBLS)
// ---------------------------------------------------------------------------
__global__ void delta_kernel(const float* __restrict__ xdbl, const float* __restrict__ dtw,
                             const float* __restrict__ dtb, float* __restrict__ delta) {
    __shared__ float sdt[DTRANK];
    const int m = blockIdx.x, d = threadIdx.x;
    if (d < DTRANK) sdt[d] = xdbl[(size_t)m * XDBLS + d];
    __syncthreads();
    float acc = dtb[d];
    #pragma unroll
    for (int r = 0; r < DTRANK; ++r) acc += sdt[r] * dtw[d * DTRANK + r];
    float sp = (acc > 20.0f) ? acc : log1pf(__expf(acc));
    delta[(size_t)m * DINNER + d] = sp;
}

// ---------------------------------------------------------------------------
// Selective scan: one wave32 per (b, d) channel, lane = state index n.
//   h[n] = exp(delta*A[n])*h[n] + (delta*u)*B_t[n];  y = sum_n h[n]*C_t[n]
// Fused output gating: yg = (y + u*Dp) * silu(z), stored f16 for out_proj GEMM.
// Latency-bound recurrence -> software prefetch (global_prefetch_b8) of t+8.
// ---------------------------------------------------------------------------
__global__ void scan_kernel(const float* __restrict__ delta, const float* __restrict__ u32,
                            const float* __restrict__ xdbl, const float* __restrict__ xz,
                            const float* __restrict__ A_log, const float* __restrict__ Dp,
                            _Float16* __restrict__ yg16) {
    const int lane = threadIdx.x & 31;
    const int wid  = blockIdx.x * (blockDim.x >> 5) + (threadIdx.x >> 5);
    const int bidx = wid >> 9;          // / 512
    const int d    = wid & (DINNER - 1);
    const float An = -__expf(A_log[d * DSTATE + lane]);
    const float Dd = Dp[d];
    float h = 0.0f;
    for (int t = 0; t < SEQL; ++t) {
        const size_t row = (size_t)(bidx << 10) + t;
        if (t + 8 < SEQL) {
            const size_t prow = row + 8;
            __builtin_prefetch(&delta[prow * DINNER + d], 0, 1);
            __builtin_prefetch(&u32[prow * DINNER + d], 0, 1);
            __builtin_prefetch(&xdbl[prow * XDBLS + DTRANK + lane], 0, 1);
            __builtin_prefetch(&xz[prow * (2 * DINNER) + DINNER + d], 0, 1);
        }
        float dlt = delta[row * DINNER + d];
        float u   = u32[row * DINNER + d];
        float Bn  = xdbl[row * XDBLS + DTRANK + lane];
        float Cn  = xdbl[row * XDBLS + DTRANK + DSTATE + lane];
        h = __expf(dlt * An) * h + (dlt * u) * Bn;
        float y = h * Cn;
        #pragma unroll
        for (int off = 16; off > 0; off >>= 1) y += __shfl_xor(y, off, 32);
        if (lane == 0) {
            float z  = xz[row * (2 * DINNER) + DINNER + d];
            float sz = z / (1.0f + __expf(-z));
            yg16[row * DINNER + d] = (_Float16)((y + u * Dd) * sz);
        }
    }
}

// ---------------------------------------------------------------------------
// WMMA GEMM: out = epilogue(A[M,K] @ W[N,K]^T). A,W f16; accumulate f32.
// Templated on K,N -> all B-tile addresses are a single base + immediate
// offsets (no per-iteration 64-bit address VALU, no WMMA WAR nops).
// Each wave: 16x64 tile (A fragment reused across 4 B fragments).
// Double-buffered k-loop: fragments for k+32 load while k feeds 4 wmmas.
// Weight buffer must have >= gridDim.x*64 rows (x_proj padded to 128).
// mode bits: 1=+bias[n], 2=exact GELU, 4=+resid[m*N+n]; outF/outH optional.
// ---------------------------------------------------------------------------
#define LOAD_FRAGS(aX, b0X, b1X, b2X, b3X, KO)                                 \
    aX.u[0]  = *(const v4u*)(Ap + (KO));                                       \
    aX.u[1]  = *(const v4u*)(Ap + (KO) + 16);                                  \
    b0X.u[0] = *(const v4u*)(Wb + 0 * 16 * K + (KO));                          \
    b0X.u[1] = *(const v4u*)(Wb + 0 * 16 * K + (KO) + 16);                     \
    b1X.u[0] = *(const v4u*)(Wb + 1 * 16 * K + (KO));                          \
    b1X.u[1] = *(const v4u*)(Wb + 1 * 16 * K + (KO) + 16);                     \
    b2X.u[0] = *(const v4u*)(Wb + 2 * 16 * K + (KO));                          \
    b2X.u[1] = *(const v4u*)(Wb + 2 * 16 * K + (KO) + 16);                     \
    b3X.u[0] = *(const v4u*)(Wb + 3 * 16 * K + (KO));                          \
    b3X.u[1] = *(const v4u*)(Wb + 3 * 16 * K + (KO) + 16);

#define WMMA4(aX, b0X, b1X, b2X, b3X)                                          \
    c0 = __builtin_amdgcn_wmma_f32_16x16x32_f16(false, aX.v, false, b0X.v,     \
                                                (short)0, c0, false, false);   \
    c1 = __builtin_amdgcn_wmma_f32_16x16x32_f16(false, aX.v, false, b1X.v,     \
                                                (short)0, c1, false, false);   \
    c2 = __builtin_amdgcn_wmma_f32_16x16x32_f16(false, aX.v, false, b2X.v,     \
                                                (short)0, c2, false, false);   \
    c3 = __builtin_amdgcn_wmma_f32_16x16x32_f16(false, aX.v, false, b3X.v,     \
                                                (short)0, c3, false, false);

template <int K, int N>
__global__ void gemm_wmma(const _Float16* __restrict__ A, const _Float16* __restrict__ W,
                          const float* __restrict__ bias, const float* __restrict__ resid,
                          float* __restrict__ outF, _Float16* __restrict__ outH,
                          int mode) {
    const int lane  = threadIdx.x & 31;
    const int wave  = threadIdx.x >> 5;
    const int mtile = blockIdx.y * 4 + wave;        // 16 rows per wave
    const int n0    = blockIdx.x * 64;              // 64 cols per wave
    const int lsel  = lane >> 4;
    const int l15   = lane & 15;
    const int mrow  = mtile * 16 + l15;
    const _Float16* Ap = A + (size_t)mrow * K + 8 * lsel;
    const _Float16* Wb = W + (size_t)(n0 + l15) * K + 8 * lsel;
    union AB { v16h v; v4u u[2]; };
    AB a0, p0, p1, p2, p3;      // buffer 0
    AB a1, q0, q1, q2, q3;      // buffer 1
    v8f c0 = {}, c1 = {}, c2 = {}, c3 = {};
    LOAD_FRAGS(a0, p0, p1, p2, p3, 0)
    int k0 = 0;
    for (; k0 < K - 64; k0 += 64) {
        LOAD_FRAGS(a1, q0, q1, q2, q3, k0 + 32)
        WMMA4(a0, p0, p1, p2, p3)
        LOAD_FRAGS(a0, p0, p1, p2, p3, k0 + 64)
        WMMA4(a1, q0, q1, q2, q3)
    }
    LOAD_FRAGS(a1, q0, q1, q2, q3, k0 + 32)
    WMMA4(a0, p0, p1, p2, p3)
    WMMA4(a1, q0, q1, q2, q3)

    const int mbase = mtile * 16 + 8 * lsel;
    #pragma unroll
    for (int j = 0; j < 4; ++j) {
        if (n0 + 16 * j >= N) continue;             // wave-uniform store guard
        const int ncol = n0 + 16 * j + l15;
        const float bval = (mode & 1) ? bias[ncol] : 0.0f;
        const v8f cc = (j == 0) ? c0 : (j == 1) ? c1 : (j == 2) ? c2 : c3;
        #pragma unroll
        for (int v = 0; v < 8; ++v) {
            float val = cc[v] + bval;
            if (mode & 2) val = 0.5f * val * (1.0f + erff(val * 0.70710678118f));
            const size_t idx = (size_t)(mbase + v) * N + ncol;
            if (mode & 4) val += resid[idx];
            if (outF) outF[idx] = val;
            if (outH) outH[idx] = (_Float16)val;
        }
    }
}

// ---------------------------------------------------------------------------
// Launch sequence
// ---------------------------------------------------------------------------
extern "C" void kernel_launch(void* const* d_in, const int* in_sizes, int n_in,
                              void* d_out, int out_size, void* d_ws, size_t ws_size,
                              hipStream_t stream) {
    const float* x        = (const float*)d_in[0];
    const float* g1       = (const float*)d_in[1];
    const float* b1       = (const float*)d_in[2];
    const float* lconv_w  = (const float*)d_in[3];
    const float* lconv_b  = (const float*)d_in[4];
    const float* in_projw = (const float*)d_in[5];
    const float* mconv_w  = (const float*)d_in[6];
    const float* mconv_b  = (const float*)d_in[7];
    const float* x_projw  = (const float*)d_in[8];
    const float* dt_w     = (const float*)d_in[9];
    const float* dt_b     = (const float*)d_in[10];
    const float* A_log    = (const float*)d_in[11];
    const float* Dp       = (const float*)d_in[12];
    const float* out_projw= (const float*)d_in[13];
    const float* g2       = (const float*)d_in[14];
    const float* b2       = (const float*)d_in[15];
    const float* w1       = (const float*)d_in[16];
    const float* bb1      = (const float*)d_in[17];
    const float* w2       = (const float*)d_in[18];
    const float* bb2      = (const float*)d_in[19];
    float* out = (float*)d_out;
    char*  ws  = (char*)d_ws;

    const size_t M = MROWS;
    // workspace layout (bytes)
    const size_t OFF_XN    = 0;                                       // f32 M*256
    const size_t OFF_U0    = OFF_XN    + M * DIMC * 4;                // f16 M*256
    const size_t OFF_WIN   = OFF_U0    + M * DIMC * 2;                // f16 1024*256
    const size_t OFF_WXP   = OFF_WIN   + (size_t)2 * DINNER * DIMC * 2;    // f16 128*512
    const size_t OFF_WOUT  = OFF_WXP   + (size_t)WXPROWS * DINNER * 2;// f16 256*512
    const size_t OFF_W1    = OFF_WOUT  + (size_t)DIMC * DINNER * 2;   // f16 1024*256
    const size_t OFF_W2    = OFF_W1    + (size_t)4 * DIMC * DIMC * 2; // f16 256*1024
    const size_t OFF_XZ    = OFF_W2    + (size_t)DIMC * 4 * DIMC * 2; // f32 M*1024
    const size_t OFF_XIN16 = OFF_XZ    + M * 2 * DINNER * 4;          // f16 M*512
    const size_t OFF_XIN32 = OFF_XIN16 + M * DINNER * 2;              // f32 M*512
    const size_t OFF_XDBL  = OFF_XIN32 + M * DINNER * 4;              // f32 M*96
    const size_t OFF_DELTA = OFF_XDBL  + M * XDBLS * 4;               // f32 M*512
    const size_t OFF_YG    = OFF_DELTA + M * DINNER * 4;              // f16 M*512
    const size_t OFF_X2    = OFF_YG    + M * DINNER * 2;              // f32 M*256
    const size_t OFF_X2LN  = OFF_X2    + M * DIMC * 4;                // f16 M*256
    const size_t OFF_A1    = OFF_X2LN  + M * DIMC * 2;                // f16 M*1024

    float*    xn    = (float*)   (ws + OFF_XN);
    _Float16* u0    = (_Float16*)(ws + OFF_U0);
    _Float16* w_in  = (_Float16*)(ws + OFF_WIN);
    _Float16* w_xp  = (_Float16*)(ws + OFF_WXP);
    _Float16* w_out = (_Float16*)(ws + OFF_WOUT);
    _Float16* w1h   = (_Float16*)(ws + OFF_W1);
    _Float16* w2h   = (_Float16*)(ws + OFF_W2);
    float*    xz    = (float*)   (ws + OFF_XZ);
    _Float16* xin16 = (_Float16*)(ws + OFF_XIN16);
    float*    xin32 = (float*)   (ws + OFF_XIN32);
    float*    xdbl  = (float*)   (ws + OFF_XDBL);
    float*    delta = (float*)   (ws + OFF_DELTA);
    _Float16* yg16  = (_Float16*)(ws + OFF_YG);
    float*    x2    = (float*)   (ws + OFF_X2);
    _Float16* x2ln  = (_Float16*)(ws + OFF_X2LN);
    _Float16* a1    = (_Float16*)(ws + OFF_A1);

    // 1. weight conversion f32 -> f16 (x_proj zero-padded to 128 rows)
    auto cvt = [&](const float* s, _Float16* d, int n) {
        cvt_f32_to_f16<<<dim3((n + 255) / 256), dim3(256), 0, stream>>>(s, d, n);
    };
    cvt(in_projw,  w_in,  2 * DINNER * DIMC);
    cvt(out_projw, w_out, DIMC * DINNER);
    cvt(w1,        w1h,   4 * DIMC * DIMC);
    cvt(w2,        w2h,   DIMC * 4 * DIMC);
    cvt_pad_xproj<<<dim3(WXPROWS * DINNER / 256), dim3(256), 0, stream>>>(x_projw, w_xp);

    // 2. LN1
    ln_kernel<<<dim3(M), dim3(DIMC), 0, stream>>>(x, g1, b1, xn, (_Float16*)nullptr);
    // 3. local conv + residual -> u0 (f16)
    conv_local_add<<<dim3(M), dim3(DIMC), 0, stream>>>(xn, lconv_w, lconv_b, u0);
    // 4. in_proj GEMM: (M,256)x(1024,256)^T -> xz (M,1024)
    gemm_wmma<DIMC, 2 * DINNER><<<dim3(2 * DINNER / 64, M / 64), dim3(128), 0, stream>>>(
        u0, w_in, nullptr, nullptr, xz, (_Float16*)nullptr, 0);
    // 5. mamba causal conv + SiLU
    conv_mamba_silu<<<dim3(M), dim3(DINNER), 0, stream>>>(xz, mconv_w, mconv_b, xin16, xin32);
    // 6. x_proj GEMM: (M,512)x(128,512)^T -> xdbl (M,96) [cols 80..95 unused]
    gemm_wmma<DINNER, XDBLS><<<dim3(WXPROWS / 64, M / 64), dim3(128), 0, stream>>>(
        xin16, w_xp, nullptr, nullptr, xdbl, (_Float16*)nullptr, 0);
    // 7. delta = softplus(dt @ dt_w.T + dt_b)
    delta_kernel<<<dim3(M), dim3(DINNER), 0, stream>>>(xdbl, dt_w, dt_b, delta);
    // 8. selective scan + gate with silu(z) -> yg16
    scan_kernel<<<dim3(BATCH * DINNER / 8), dim3(256), 0, stream>>>(
        delta, xin32, xdbl, xz, A_log, Dp, yg16);
    // 9. out_proj GEMM + residual x -> x2 (M,256)
    gemm_wmma<DINNER, DIMC><<<dim3(DIMC / 64, M / 64), dim3(128), 0, stream>>>(
        yg16, w_out, nullptr, x, x2, (_Float16*)nullptr, 4);
    // 10. LN2 -> x2ln (f16)
    ln_kernel<<<dim3(M), dim3(DIMC), 0, stream>>>(x2, g2, b2, (float*)nullptr, x2ln);
    // 11. MLP1: bias + exact GELU -> a1 (f16)
    gemm_wmma<DIMC, 4 * DIMC><<<dim3(4 * DIMC / 64, M / 64), dim3(128), 0, stream>>>(
        x2ln, w1h, bb1, nullptr, (float*)nullptr, a1, 1 | 2);
    // 12. MLP2: bias + residual x2 -> d_out
    gemm_wmma<4 * DIMC, DIMC><<<dim3(DIMC / 64, M / 64), dim3(128), 0, stream>>>(
        a1, w2h, bb2, x2, out, (_Float16*)nullptr, 1 | 4);
    (void)in_sizes; (void)n_in; (void)out_size; (void)ws_size;
}
